// CausalAttention_81544249082055
// MI455X (gfx1250) — compile-verified, matrix operand order checked by reading, not run on hardware
//
#include <hip/hip_runtime.h>

// Causal (PixelSNAIL) attention, flash-style, bf16 WMMA + TDM async staging
// on gfx1250. B=4, S=4096, D=256. Grid: 4*64 workgroups of 128 threads
// (4 wave32's). Each wave owns 16 query rows; block: 64 query rows x full D.

#define SLEN  4096
#define DIM   256
#define BM    64
#define BN    64
#define STRQK 264   // DIM + 8 bf16 pad (16B) -> bank spread
#define STRV  72    // BN  + 8 bf16 pad
#define NEG_BIG 1e9f

typedef __attribute__((ext_vector_type(16))) __bf16 v16bf;
typedef __attribute__((ext_vector_type(8)))  float  v8f;
typedef __attribute__((ext_vector_type(4)))  unsigned int v4u;
typedef __attribute__((ext_vector_type(8)))  int v8i;
typedef __attribute__((ext_vector_type(4)))  int v4i;

union FragBF {
  uint4 q[2];
  v16bf v;
};

static __device__ __forceinline__ unsigned short f2bf(float f) {
  unsigned u = __builtin_bit_cast(unsigned, f);
  unsigned r = (u + 0x7FFFu + ((u >> 16) & 1u)) >> 16;  // RNE
  return (unsigned short)r;
}

// Issue a TDM 2D tile load: rows x 256 fp32, row-major, global -> LDS.
// Descriptor per CDNA5 ISA (D# groups).
static __device__ __forceinline__ void tdm_load_tile(const float* gsrc,
                                                     unsigned lds_byte_off) {
  unsigned long long ga = (unsigned long long)(uintptr_t)gsrc;
  v4u g0;
  g0.x = 1u;                                           // count=1, user D#
  g0.y = lds_byte_off;                                 // lds_addr (bytes)
  g0.z = (unsigned)(ga & 0xFFFFFFFFu);                 // global_addr[31:0]
  g0.w = (unsigned)((ga >> 32) & 0x1FFFFFFu) | (2u << 30);  // [56:32] | type=2
  v8i g1;
  g1[0] = (int)(2u << 16);        // workgroup_mask=0, data_size=4B (code 2)
  g1[1] = (int)(256u << 16);      // atomic_barrier=0 | tensor_dim0[15:0]=256
  g1[2] = (int)(4096u << 16);     // tensor_dim0[31:16]=0 | tensor_dim1[15:0]=4096
  g1[3] = (int)(256u << 16);      // tensor_dim1[31:16]=0 | tile_dim0=256
  g1[4] = BN;                     // tile_dim1=64, tile_dim2=0
  g1[5] = 256;                    // tensor_dim0_stride = 256 elements
  g1[6] = 0;                      // stride0 hi | tensor_dim1_stride lo
  g1[7] = 0;                      // tensor_dim1_stride hi
  v4i z4 = {0, 0, 0, 0};
#if __clang_major__ >= 23
  v8i z8 = {0, 0, 0, 0, 0, 0, 0, 0};
  __builtin_amdgcn_tensor_load_to_lds(g0, g1, z4, z4, z8, 0);
#else
  __builtin_amdgcn_tensor_load_to_lds(g0, g1, z4, z4, 0);
#endif
}

__global__ void __launch_bounds__(128, 1)
causal_attn_wmma_kernel(const float* __restrict__ Q,
                        const float* __restrict__ K,
                        const float* __restrict__ V,
                        float* __restrict__ O) {
  extern __shared__ unsigned short smem[];
  unsigned short* ldQ  = smem;                    // [BM][STRQK] bf16
  unsigned short* ldK  = ldQ  + BM  * STRQK;      // [BN][STRQK] bf16
  unsigned short* ldVt = ldK  + BN  * STRQK;      // [DIM][STRV] bf16 (V^T)
  unsigned short* ldP  = ldVt + DIM * STRV;       // [BM][STRV]  bf16 (per-wave)
  // fp32 staging for TDM (raw tiles, converted after arrival)
  const unsigned stK_off = (unsigned)((BM * STRQK + BN * STRQK + DIM * STRV + BM * STRV) *
                                      sizeof(unsigned short));
  const unsigned stV_off = stK_off + BN * DIM * sizeof(float);
  float* stK = (float*)((char*)smem + stK_off);   // [BN][DIM] fp32
  float* stV = (float*)((char*)smem + stV_off);   // [BN][DIM] fp32

  const int tid  = threadIdx.x;
  const int lane = tid & 31;
  const int wave = tid >> 5;
  const int r16  = lane & 15;
  const int half = lane >> 4;

  const int qt = blockIdx.x & 63;
  const int b  = blockIdx.x >> 6;
  const int q0 = qt * BM;

  const float* Qb = Q + (size_t)b * SLEN * DIM;
  const float* Kb = K + (size_t)b * SLEN * DIM;
  const float* Vb = V + (size_t)b * SLEN * DIM;
  float*       Ob = O + (size_t)b * SLEN * DIM;

  // LDS base byte offset of our dynamic allocation (flat addr low 32 bits
  // are the LDS offset per the CDNA5 aperture mapping).
  const unsigned ldsBase = (unsigned)(uintptr_t)(void*)smem;

  // ---- kick off TDM for key/value tile 0 while we stage Q ----
  if (wave == 0) {
    tdm_load_tile(Kb, ldsBase + stK_off);
    tdm_load_tile(Vb, ldsBase + stV_off);
  }

  // ---- stage Q tile once: 64x256 f32 -> bf16 row-major LDS ----
  for (int i = 0; i < 32; ++i) {
    int f   = tid + i * 128;            // float4 index within tile (64*64)
    int row = f >> 6;                   // 64 float4 per row
    int c4  = (f & 63) << 2;
    float4 d = *(const float4*)(Qb + (size_t)(q0 + row) * DIM + c4);
    unsigned short h0 = f2bf(d.x), h1 = f2bf(d.y), h2 = f2bf(d.z), h3 = f2bf(d.w);
    *(uint2*)(ldQ + row * STRQK + c4) =
        make_uint2((unsigned)h0 | ((unsigned)h1 << 16),
                   (unsigned)h2 | ((unsigned)h3 << 16));
  }

  const float scale = 0.0625f;  // 1/sqrt(256)

  v8f oAcc[16];
  #pragma unroll
  for (int i = 0; i < 16; ++i) oAcc[i] = {};
  float Mrow[8], Lrow[8];
  #pragma unroll
  for (int i = 0; i < 8; ++i) { Mrow[i] = -__builtin_inff(); Lrow[i] = 0.f; }

  const int nJ = qt + 1;  // causal: only tiles with keys j < q0+BM
  for (int jt = 0; jt < nJ; ++jt) {
    const int j0 = jt * BN;

    // TDM data for this tile has landed once issuing wave drains TENSORcnt;
    // barrier also guarantees all waves finished reading ldK/ldVt of tile-1.
    if (wave == 0) __builtin_amdgcn_s_wait_tensorcnt(0);
    __syncthreads();

    // ---- convert staged K tile: fp32 LDS -> bf16 row-major LDS ----
    for (int i = 0; i < 32; ++i) {
      int f   = tid + i * 128;
      int row = f >> 6;
      int c4  = (f & 63) << 2;
      float4 d = *(const float4*)(stK + row * DIM + c4);
      unsigned short h0 = f2bf(d.x), h1 = f2bf(d.y), h2 = f2bf(d.z), h3 = f2bf(d.w);
      *(uint2*)(ldK + row * STRQK + c4) =
          make_uint2((unsigned)h0 | ((unsigned)h1 << 16),
                     (unsigned)h2 | ((unsigned)h3 << 16));
    }
    // ---- convert staged V tile transposed: ldVt[c][j] ----
    for (int i = 0; i < 32; ++i) {
      int f   = tid + i * 128;
      int row = f >> 6;                 // j within tile
      int c4  = (f & 63) << 2;
      float4 d = *(const float4*)(stV + row * DIM + c4);
      ldVt[(c4 + 0) * STRV + row] = f2bf(d.x);
      ldVt[(c4 + 1) * STRV + row] = f2bf(d.y);
      ldVt[(c4 + 2) * STRV + row] = f2bf(d.z);
      ldVt[(c4 + 3) * STRV + row] = f2bf(d.w);
    }
    __syncthreads();  // bf16 buffers ready; staging buffers now free

    // ---- issue TDM for the next tile; streams in behind the WMMAs ----
    if (jt + 1 < nJ && wave == 0) {
      tdm_load_tile(Kb + (size_t)(j0 + BN) * DIM, ldsBase + stK_off);
      tdm_load_tile(Vb + (size_t)(j0 + BN) * DIM, ldsBase + stV_off);
    }

    // ---- S = Q K^T : this wave's 16 rows x 64 cols, 4 accumulators ----
    v8f sAcc[4];
    #pragma unroll
    for (int nt = 0; nt < 4; ++nt) {
      v8f c = {};
      #pragma unroll
      for (int kc = 0; kc < 8; ++kc) {
        FragBF a, bf;
        // A frag: rows of Q (lane r16), K-halves split by lane>=16
        const unsigned short* pa = ldQ + (wave * 16 + r16) * STRQK + kc * 32 + half * 8;
        a.q[0] = *(const uint4*)pa;
        a.q[1] = *(const uint4*)(pa + 16);
        // B frag for Q*K^T: B[k][n] = K[n][k] -> row-major K rows, mirror layout
        const unsigned short* pb = ldK + (nt * 16 + r16) * STRQK + kc * 32 + half * 8;
        bf.q[0] = *(const uint4*)pb;
        bf.q[1] = *(const uint4*)(pb + 16);
        c = __builtin_amdgcn_wmma_f32_16x16x32_bf16(false, a.v, false, bf.v,
                                                    (short)0, c, false, false);
      }
      sAcc[nt] = c;
    }

    // ---- online softmax (C layout: lane = col r16, row = r + 8*half) ----
    #pragma unroll
    for (int r = 0; r < 8; ++r) {
      const int rowg = q0 + wave * 16 + r + 8 * half;
      float mx = -__builtin_inff();
      #pragma unroll
      for (int nt = 0; nt < 4; ++nt) {
        const int colg = j0 + nt * 16 + r16;
        float s = sAcc[nt][r] * scale;
        if (colg >= rowg) s -= NEG_BIG;   // reference: scores - NEG_BIG*(1-allowed)
        sAcc[nt][r] = s;
        mx = fmaxf(mx, s);
      }
      // reduce across the 16 lanes that hold this row
      mx = fmaxf(mx, __shfl_xor(mx, 1, 32));
      mx = fmaxf(mx, __shfl_xor(mx, 2, 32));
      mx = fmaxf(mx, __shfl_xor(mx, 4, 32));
      mx = fmaxf(mx, __shfl_xor(mx, 8, 32));
      const float mnew  = fmaxf(Mrow[r], mx);
      const float alpha = __expf(Mrow[r] - mnew);
      Mrow[r] = mnew;
      float rs = 0.f;
      #pragma unroll
      for (int nt = 0; nt < 4; ++nt) {
        const int colg = j0 + nt * 16 + r16;
        // reference multiplies probs by `allowed` -> masked prob exactly 0
        float p = (colg < rowg) ? __expf(sAcc[nt][r] - mnew) : 0.f;
        rs += p;
        ldP[(wave * 16 + r + 8 * half) * STRV + nt * 16 + r16] = f2bf(p);
      }
      rs += __shfl_xor(rs, 1, 32);
      rs += __shfl_xor(rs, 2, 32);
      rs += __shfl_xor(rs, 4, 32);
      rs += __shfl_xor(rs, 8, 32);
      Lrow[r] = Lrow[r] * alpha + rs;
      #pragma unroll
      for (int ct = 0; ct < 16; ++ct) oAcc[ct][r] *= alpha;
    }

    // ---- O += P * V (P from per-wave LDS slab; V^T gives B frags directly) ----
    #pragma unroll
    for (int ct = 0; ct < 16; ++ct) {
      v8f c = oAcc[ct];
      #pragma unroll
      for (int kc = 0; kc < 2; ++kc) {
        FragBF a, bf;
        const unsigned short* pa = ldP + (wave * 16 + r16) * STRV + kc * 32 + half * 8;
        a.q[0] = *(const uint4*)pa;
        a.q[1] = *(const uint4*)(pa + 16);
        const unsigned short* pb = ldVt + (ct * 16 + r16) * STRV + kc * 32 + half * 8;
        bf.q[0] = *(const uint4*)pb;
        bf.q[1] = *(const uint4*)(pb + 16);
        c = __builtin_amdgcn_wmma_f32_16x16x32_bf16(false, a.v, false, bf.v,
                                                    (short)0, c, false, false);
      }
      oAcc[ct] = c;
    }
  }

  // ---- epilogue: O / L ; rows with L==0 (global row 0) -> zeros ----
  #pragma unroll
  for (int r = 0; r < 8; ++r) {
    const int rowg = q0 + wave * 16 + r + 8 * half;
    const float inv = (Lrow[r] > 0.f) ? 1.0f / Lrow[r] : 0.0f;
    #pragma unroll
    for (int ct = 0; ct < 16; ++ct) {
      Ob[(size_t)rowg * DIM + ct * 16 + r16] = oAcc[ct][r] * inv;
    }
  }
}

extern "C" void kernel_launch(void* const* d_in, const int* in_sizes, int n_in,
                              void* d_out, int out_size, void* d_ws, size_t ws_size,
                              hipStream_t stream) {
  (void)in_sizes; (void)n_in; (void)out_size; (void)d_ws; (void)ws_size;
  const float* Q = (const float*)d_in[0];
  const float* K = (const float*)d_in[1];
  const float* V = (const float*)d_in[2];
  float*       O = (float*)d_out;

  const size_t bf16_bytes =
      (size_t)(BM * STRQK + BN * STRQK + DIM * STRV + BM * STRV) * sizeof(unsigned short);
  const size_t shmem = bf16_bytes + 2 * (size_t)BN * DIM * sizeof(float);
  dim3 grid(4 * (SLEN / BM));  // 256 workgroups
  dim3 block(128);             // 4 waves (wave32)
  hipLaunchKernelGGL(causal_attn_wmma_kernel, grid, block, shmem, stream, Q, K, V, O);
}
